// SheafBuilderLowRank_74509092651432
// MI455X (gfx1250) — matrix operand order repaired and checked by reading, compiled, not run on hardware
//
#include <hip/hip_runtime.h>
#include <hip/hip_bf16.h>

typedef _Float16 v16h __attribute__((ext_vector_type(16)));
typedef _Float16 v8h  __attribute__((ext_vector_type(8)));
typedef _Float16 v4h  __attribute__((ext_vector_type(4)));
typedef float    v8f  __attribute__((ext_vector_type(8)));

#define DDIM 6
#define FDIM 64
#define OUTC 30            // 2*D*RANK + D = 30
#define WAVES_PER_BLOCK 8

// ---------------------------------------------------------------------------
// Kernel 0: pre-pack W (128x30 f32) into the WMMA B-operand lane layout as
// f16, once for the whole dispatch. Layout: op = tile*4 + c (tile = N half,
// c = K chunk); element [(op*32 + lane)*16 + t] = W[32c + (lane>>4)*16 + t]
// [ (lane&15) + 16*tile ], zero-padded for n >= 30.  8 KB total.
// ---------------------------------------------------------------------------
__global__ void pack_W_kernel(const float* __restrict__ Wmat,
                              _Float16* __restrict__ wpack) {
  const int tid  = threadIdx.x;          // one block of 256 = 8 ops x 32 lanes
  const int op   = tid >> 5;
  const int lane = tid & 31;
  const int tile = op >> 2;
  const int c    = op & 3;
  const int n    = (lane & 15) + 16 * tile;
  const int kb   = 32 * c + ((lane >> 4) << 4);
  v16h v;
#pragma unroll
  for (int t = 0; t < 16; ++t)
    v[t] = (n < OUTC) ? (_Float16)Wmat[(kb + t) * OUTC + n] : (_Float16)0.0f;
  *(v16h*)(wpack + (size_t)(op * 32 + lane) * 16) = v;
}

// ---------------------------------------------------------------------------
// Kernel 1: mean over the D=6 axis, store as f16 (WMMA A-operand dtype).
// ---------------------------------------------------------------------------
__global__ void __launch_bounds__(256)
mean6_to_f16(const float* __restrict__ in, _Float16* __restrict__ out, int rows) {
  int t = blockIdx.x * blockDim.x + threadIdx.x;
  if (t >= rows * 16) return;
  int r = t >> 4;
  int q = (t & 15) << 2;
  const float* p = in + (size_t)r * (DDIM * FDIM) + q;
  float s0 = 0.f, s1 = 0.f, s2 = 0.f, s3 = 0.f;
#pragma unroll
  for (int d = 0; d < DDIM; ++d) {
    const float4 v = *(const float4*)(p + d * FDIM);
    s0 += v.x; s1 += v.y; s2 += v.z; s3 += v.w;
  }
  const float sc = 1.0f / 6.0f;
  v4h h;
  h[0] = (_Float16)(s0 * sc);
  h[1] = (_Float16)(s1 * sc);
  h[2] = (_Float16)(s2 * sc);
  h[3] = (_Float16)(s3 * sc);
  *(v4h*)(out + (size_t)r * FDIM + q) = h;
}

// ---------------------------------------------------------------------------
// Kernel 2: one wave32 handles 16 (row,col) pairs.
//   A (16x128 f16) x W (128x30 f16) via 2 tiles x 4 chained
//   v_wmma_f32_16x16x32_f16, sigmoid, then per-pair 6x6 low-rank map
//   A6x2 * B6x2^T + diag(C). Output stores are coalesced and non-temporal
//   (write-once streams; keep L2 for the mean-table / index gathers).
// ---------------------------------------------------------------------------
__global__ void __launch_bounds__(256)
sheaf_main(const _Float16* __restrict__ xm, const _Float16* __restrict__ em,
           const _Float16* __restrict__ wpack, const float* __restrict__ bias,
           const long long* __restrict__ he, long long nnz,
           long long* __restrict__ idx0, long long* __restrict__ idx1,
           float* __restrict__ attrs)
{
  __shared__ float     llh[WAVES_PER_BLOCK][16][33];   // per-wave h_sheaf tile
  __shared__ long long lrc[WAVES_PER_BLOCK][2][16];    // per-wave row/col cache

  const int lane  = threadIdx.x & 31;
  const int wv    = threadIdx.x >> 5;
  const long long numGroups = (nnz + 15) >> 4;
  const long long g = (long long)blockIdx.x * WAVES_PER_BLOCK + wv;
  if (g >= numGroups) return;          // wave-uniform: EXEC stays all-ones

  const int n0    = lane & 15;         // output column (tile 0) / A row M
  const int halfo = lane >> 4;         // lane half selects K sub-range

  // ---- load pre-packed B operands: 8 x v16h, coalesced b128 loads ----
  const v16h* wp = (const v16h*)wpack;
  v16h Bop[2][4];
#pragma unroll
  for (int tile = 0; tile < 2; ++tile)
#pragma unroll
    for (int c = 0; c < 4; ++c)
      Bop[tile][c] = wp[(((tile << 2) + c) << 5) + lane];

  const float b0 = bias[n0];
  const float b1 = (n0 + 16 < OUTC) ? bias[n0 + 16] : 0.0f;

  // ---- fetch pair indices; lane's A-rows belong to pair M = lane&15 ----
  const long long pM  = g * 16 + n0;
  const long long pc  = (pM < nnz) ? pM : (nnz - 1);
  const long long row = he[pc];
  const long long col = he[nnz + pc];
  if (lane < 16) { lrc[wv][0][lane] = row; lrc[wv][1][lane] = col; }

  const _Float16* xrow = xm + (size_t)row * FDIM;
  const _Float16* erow = em + (size_t)col * FDIM;

  // ---- 16x128 @ 128x30 via WMMA: chunks c=0,1 from x-half, c=2,3 from e ----
  // A operand (16x32 f16): lane row M = lane&15; halves 0..15 = K kb..kb+15,
  // kb = 32c + halfo*16.
  v8f acc0 = {};
  v8f acc1 = {};
#pragma unroll
  for (int c = 0; c < 4; ++c) {
    const _Float16* src = (c < 2) ? xrow : erow;
    const int o1 = ((c & 1) << 5) + (halfo << 4);
    v8h lo = *(const v8h*)(src + o1);        // merges into global_load_b128
    v8h hi = *(const v8h*)(src + o1 + 8);
    v16h a = __builtin_shufflevector(lo, hi,
        0, 1, 2, 3, 4, 5, 6, 7, 8, 9, 10, 11, 12, 13, 14, 15);
    acc0 = __builtin_amdgcn_wmma_f32_16x16x32_f16(
        false, a, false, Bop[0][c], (short)0, acc0, false, false);
    acc1 = __builtin_amdgcn_wmma_f32_16x16x32_f16(
        false, a, false, Bop[1][c], (short)0, acc1, false, false);
  }

  // ---- bias + sigmoid, stash 16x30 tile into per-wave LDS ----
  // C/D layout: VGPR r, lanes 0-15 -> M=r, lanes 16-31 -> M=r+8; N = lane&15.
  const int Mb = halfo << 3;
#pragma unroll
  for (int r = 0; r < 8; ++r) {
    float s0 = acc0[r] + b0;
    float s1 = acc1[r] + b1;
    s0 = 1.0f / (1.0f + __expf(-s0));
    s1 = 1.0f / (1.0f + __expf(-s1));
    llh[wv][Mb + r][n0]      = s0;
    llh[wv][Mb + r][n0 + 16] = s1;
  }
  // same-wave DS RAW: wait for all LDS writes before cross-lane reads
  asm volatile("s_wait_dscnt 0" ::: "memory");

  // ---- maps = A(6x2) * B(6x2)^T + diag(C); coalesced NT output stores ----
  const long long gb   = g * 576;          // 16 pairs * 36 entries
  const bool      full = (g * 16 + 16) <= nnz;
#pragma unroll 2
  for (int u = 0; u < 18; ++u) {
    const int t    = u * 32 + lane;        // 0..575, coalesced across lanes
    const int pair = t / 36;
    const int k    = t - pair * 36;
    const int i    = k / 6;
    const int j    = k - i * 6;
    const float* hp = &llh[wv][pair][0];
    float v = hp[2 * i] * hp[12 + 2 * j] + hp[2 * i + 1] * hp[13 + 2 * j];
    if (i == j) v += hp[24 + i];
    if (full || (g * 16 + pair) < nnz) {
      __builtin_nontemporal_store(v, attrs + gb + t);
      __builtin_nontemporal_store(6LL * lrc[wv][0][pair] + i, idx0 + gb + t);
      __builtin_nontemporal_store(6LL * lrc[wv][1][pair] + j, idx1 + gb + t);
    }
  }
}

// ---------------------------------------------------------------------------
extern "C" void kernel_launch(void* const* d_in, const int* in_sizes, int n_in,
                              void* d_out, int out_size, void* d_ws, size_t ws_size,
                              hipStream_t stream) {
  const float*     x  = (const float*)d_in[0];
  const float*     e  = (const float*)d_in[1];
  const float*     Wm = (const float*)d_in[2];
  const float*     b  = (const float*)d_in[3];
  const long long* he = (const long long*)d_in[4];   // int64 (2, NNZ)

  const long long N   = (long long)in_sizes[0] / (DDIM * FDIM);
  const long long E   = (long long)in_sizes[1] / (DDIM * FDIM);
  const long long NNZ = (long long)in_sizes[4] / 2;

  // workspace: packed W (8 KB) + f16 mean tables (N*64 + E*64 halves)
  _Float16* wpack = (_Float16*)d_ws;
  _Float16* xm    = wpack + 8 * 32 * 16;
  _Float16* em    = xm + (size_t)N * FDIM;

  pack_W_kernel<<<1, 256, 0, stream>>>(Wm, wpack);

  {
    const long long tx = N * 16;
    mean6_to_f16<<<dim3((unsigned)((tx + 255) / 256)), 256, 0, stream>>>(x, xm, (int)N);
    const long long te = E * 16;
    mean6_to_f16<<<dim3((unsigned)((te + 255) / 256)), 256, 0, stream>>>(e, em, (int)E);
  }

  // tuple output, flat in return order:
  //   h_sheaf_index (2, NNZ*36) int64  -> idx0 then idx1
  //   h_attrs       (NNZ*36)    float32
  long long* idx0  = (long long*)d_out;
  long long* idx1  = idx0 + NNZ * 36;
  float*     attrs = (float*)(idx1 + NNZ * 36);

  const long long groups = (NNZ + 15) / 16;
  const long long blocks = (groups + WAVES_PER_BLOCK - 1) / WAVES_PER_BLOCK;
  sheaf_main<<<dim3((unsigned)blocks), 32 * WAVES_PER_BLOCK, 0, stream>>>(
      xm, em, wpack, b, he, NNZ, idx0, idx1, attrs);
}
